// NetBinary_14937896256083
// MI455X (gfx1250) — 8-bit matrix operand order repaired and checked by reading, compiled, not run on hardware
//
#include <hip/hip_runtime.h>
#include <stdint.h>


typedef __attribute__((ext_vector_type(8))) int   v8i;
typedef __attribute__((ext_vector_type(4))) int   v4i;
typedef __attribute__((ext_vector_type(2))) int   v2i;
typedef __attribute__((ext_vector_type(4))) float v4f;

#define BN_EPS 1e-5f

__device__ __forceinline__ int sgn_i(float x) {
    return x > 0.f ? 1 : (x < 0.f ? -1 : 0);
}
__device__ __forceinline__ int pack4sgn(float a, float b, float c, float d) {
    return (sgn_i(a) & 0xFF) | ((sgn_i(b) & 0xFF) << 8) |
           ((sgn_i(c) & 0xFF) << 16) | ((sgn_i(d) & 0xFF) << 24);
}

__global__ __launch_bounds__(256) void k_binw1(const float* __restrict__ cw1,
                                               int8_t* __restrict__ bw1) {
    int idx = blockIdx.x * 256 + threadIdx.x;
    if (idx >= 128 * 576) return;
    int co = idx / 576, r = idx % 576;
    int t = r / 64, ci = r % 64;
    int dy = t / 3, dx = t % 3;
    float w = cw1[((co * 64 + ci) * 3 + dy) * 3 + dx];
    bw1[idx] = (int8_t)sgn_i(w);
}

__global__ __launch_bounds__(256) void k_conv0(const float* __restrict__ x,
                                               const float* __restrict__ cw0,
                                               int8_t* __restrict__ t0) {
    int m  = blockIdx.x * 256 + threadIdx.x;
    int co = blockIdx.y;
    int n = m / 784, hw = m % 784, h = hw / 28, w = hw % 28;
    int acc = 0;
#pragma unroll
    for (int dy = 0; dy < 3; ++dy)
#pragma unroll
        for (int dx = 0; dx < 3; ++dx) {
            int hh = h + dy - 1, ww = w + dx - 1;
            if (hh >= 0 && hh < 28 && ww >= 0 && ww < 28) {
                acc += sgn_i(x[n * 784 + hh * 28 + ww]) *
                       sgn_i(cw0[co * 9 + dy * 3 + dx]);
            }
        }
    t0[(long)co * 200704 + m] = (int8_t)acc;
}

__global__ __launch_bounds__(256) void k_bnbin0(const int8_t* __restrict__ t0,
                                                const float* __restrict__ p,
                                                const float* __restrict__ g,
                                                const float* __restrict__ b,
                                                int8_t* __restrict__ a0p) {
    const int M = 200704;
    int c = blockIdx.x, tid = threadIdx.x;
    float slope = p[c];
    float s = 0.f, s2 = 0.f;
    for (int i = tid; i < M; i += 256) {
        float v = (float)t0[(long)c * M + i];
        float u = v >= 0.f ? v : slope * v;
        s += u; s2 += u * u;
    }
    __shared__ float sh[256], sh2[256];
    sh[tid] = s; sh2[tid] = s2; __syncthreads();
    for (int o = 128; o > 0; o >>= 1) {
        if (tid < o) { sh[tid] += sh[tid + o]; sh2[tid] += sh2[tid + o]; }
        __syncthreads();
    }
    float mean = sh[0] / (float)M;
    float var  = sh2[0] / (float)M - mean * mean;
    float inv  = rsqrtf(var + BN_EPS) * g[c];
    float beta = b[c];
    for (int i = tid; i < M; i += 256) {
        float v = (float)t0[(long)c * M + i];
        float u = v >= 0.f ? v : slope * v;
        float y = (u - mean) * inv + beta;
        int n = i / 784, hw = i % 784, h = hw / 28, w = hw % 28;
        long pos = (long)n * 900 + (h + 1) * 30 + (w + 1);
        a0p[pos * 64 + c] = (int8_t)sgn_i(y);
    }
}

__global__ __launch_bounds__(256) void k_conv1(const int8_t* __restrict__ a0p,
                                               const int8_t* __restrict__ bw1,
                                               int16_t* __restrict__ t1) {
    int lane  = threadIdx.x & 31;
    int wv    = threadIdx.x >> 5;
    int mtile = blockIdx.x;
    int lm    = lane & 15;
    int half  = lane >> 4;
    int m = mtile * 16 + lm;
    int n = m / 784, hw = m % 784, h = hw / 28, w = hw % 28;
    int co = wv * 16 + lm;
    long posbase = (long)n * 900 + h * 30 + w;

    v8i acc = {};
#pragma unroll
    for (int t = 0; t < 9; ++t) {
        int dy = t / 3, dx = t % 3;
        const int8_t* ap = a0p + (posbase + dy * 30 + dx) * 64;
        v8i A, B;
#pragma unroll
        for (int j = 0; j < 4; ++j) {
            v2i av = *(const v2i*)(ap + (j >> 1) * 32 + half * 16 + (j & 1) * 8);
            A[2 * j] = av.x; A[2 * j + 1] = av.y;
        }
        const int8_t* bp = bw1 + (long)co * 576 + t * 64 + half * 16;
        v4i b0 = *(const v4i*)(bp);
        v4i b1 = *(const v4i*)(bp + 32);
        B[0] = b0.x; B[1] = b0.y; B[2] = b0.z; B[3] = b0.w;
        B[4] = b1.x; B[5] = b1.y; B[6] = b1.z; B[7] = b1.w;
        acc = __builtin_amdgcn_wmma_i32_16x16x64_iu8(true, A, true, B, acc,
                                                     false, false);
    }
    int mbase = mtile * 16 + half * 8;
    v4i pk;
    pk.x = (acc[0] & 0xFFFF) | (acc[1] << 16);
    pk.y = (acc[2] & 0xFFFF) | (acc[3] << 16);
    pk.z = (acc[4] & 0xFFFF) | (acc[5] << 16);
    pk.w = (acc[6] & 0xFFFF) | (acc[7] << 16);
    *(v4i*)(t1 + (long)co * 200704 + mbase) = pk;
}

__global__ __launch_bounds__(256) void k_bnbin1(const int16_t* __restrict__ t1,
                                                const float* __restrict__ p,
                                                const float* __restrict__ g,
                                                const float* __restrict__ b,
                                                int8_t* __restrict__ a1) {
    const int M = 200704;
    int c = blockIdx.x, tid = threadIdx.x;
    float slope = p[c];
    float s = 0.f, s2 = 0.f;
    for (int i = tid; i < M; i += 256) {
        float v = (float)t1[(long)c * M + i];
        float u = v >= 0.f ? v : slope * v;
        s += u; s2 += u * u;
    }
    __shared__ float sh[256], sh2[256];
    sh[tid] = s; sh2[tid] = s2; __syncthreads();
    for (int o = 128; o > 0; o >>= 1) {
        if (tid < o) { sh[tid] += sh[tid + o]; sh2[tid] += sh2[tid + o]; }
        __syncthreads();
    }
    float mean = sh[0] / (float)M;
    float var  = sh2[0] / (float)M - mean * mean;
    float inv  = rsqrtf(var + BN_EPS) * g[c];
    float beta = b[c];
    for (int i = tid; i < M; i += 256) {
        float v = (float)t1[(long)c * M + i];
        float u = v >= 0.f ? v : slope * v;
        float y = (u - mean) * inv + beta;
        int n = i / 784, hw = i % 784;
        a1[(long)n * 100352 + c * 784 + hw] = (int8_t)sgn_i(y);
    }
}

__global__ __launch_bounds__(256) void k_fc0(const int8_t* __restrict__ a1,
                                             const float* __restrict__ fw0,
                                             int* __restrict__ tfc0) {
    int lane = threadIdx.x & 31;
    int wv   = threadIdx.x >> 5;
    int lm   = lane & 15;
    int half = lane >> 4;
    int ks    = blockIdx.x;
    int ntile = blockIdx.y * 8 + wv;
    int f     = ntile * 16 + lm;

    v8i acc[16];
#pragma unroll
    for (int i = 0; i < 16; ++i) acc[i] = (v8i){};

    const float* wrow = fw0 + (long)f * 100352;
    for (int kc = 0; kc < 49; ++kc) {
        int k0 = (ks * 49 + kc) * 64;
        __builtin_prefetch(wrow + k0 + 64, 0, 1);
        const float* wpl = wrow + k0 + half * 16;
        v8i B;
#pragma unroll
        for (int i = 0; i < 4; ++i) {
            v4f w0 = *(const v4f*)(wpl + 4 * i);
            v4f w1 = *(const v4f*)(wpl + 32 + 4 * i);
            B[i]     = pack4sgn(w0.x, w0.y, w0.z, w0.w);
            B[i + 4] = pack4sgn(w1.x, w1.y, w1.z, w1.w);
        }
#pragma unroll
        for (int mt = 0; mt < 16; ++mt) {
            int m = mt * 16 + lm;
            const int8_t* ap = a1 + (long)m * 100352 + k0;
            v8i A;
#pragma unroll
            for (int j = 0; j < 4; ++j) {
                v2i av = *(const v2i*)(ap + (j >> 1) * 32 + half * 16 + (j & 1) * 8);
                A[2 * j] = av.x; A[2 * j + 1] = av.y;
            }
            acc[mt] = __builtin_amdgcn_wmma_i32_16x16x64_iu8(true, A, true, B,
                                                             acc[mt], false,
                                                             false);
        }
    }
#pragma unroll
    for (int mt = 0; mt < 16; ++mt) {
        int mbase = mt * 16 + half * 8;
#pragma unroll
        for (int r = 0; r < 8; ++r)
            atomicAdd(&tfc0[f * 256 + mbase + r], acc[mt][r]);
    }
}

__global__ __launch_bounds__(256) void k_bnfc0(const int* __restrict__ tfc0,
                                               const float* __restrict__ p,
                                               const float* __restrict__ g,
                                               const float* __restrict__ b,
                                               int8_t* __restrict__ a2) {
    int f = blockIdx.x, tid = threadIdx.x;
    float slope = p[f];
    float v = (float)tfc0[f * 256 + tid];
    float u = v >= 0.f ? v : slope * v;
    __shared__ float sh[256], sh2[256];
    sh[tid] = u; sh2[tid] = u * u; __syncthreads();
    for (int o = 128; o > 0; o >>= 1) {
        if (tid < o) { sh[tid] += sh[tid + o]; sh2[tid] += sh2[tid + o]; }
        __syncthreads();
    }
    float mean = sh[0] / 256.f;
    float var  = sh2[0] / 256.f - mean * mean;
    float y = (u - mean) * rsqrtf(var + BN_EPS) * g[f] + b[f];
    a2[(long)tid * 1024 + f] = (int8_t)sgn_i(y);
}

__global__ __launch_bounds__(256) void k_fc1(const int8_t* __restrict__ a2,
                                             const float* __restrict__ fw1,
                                             const float* __restrict__ p,
                                             const float* __restrict__ g,
                                             const float* __restrict__ b,
                                             const float* __restrict__ scale,
                                             float* __restrict__ out) {
    int f = blockIdx.x, m = threadIdx.x;
    int acc = 0;
    for (int k = 0; k < 1024; ++k)
        acc += (int)a2[(long)m * 1024 + k] * sgn_i(fw1[f * 1024 + k]);
    float slope = p[f];
    float u = acc >= 0 ? (float)acc : slope * (float)acc;
    __shared__ float sh[256], sh2[256];
    sh[m] = u; sh2[m] = u * u; __syncthreads();
    for (int o = 128; o > 0; o >>= 1) {
        if (m < o) { sh[m] += sh[m + o]; sh2[m] += sh2[m + o]; }
        __syncthreads();
    }
    float mean = sh[0] / 256.f;
    float var  = sh2[0] / 256.f - mean * mean;
    float y = (u - mean) * rsqrtf(var + BN_EPS) * g[f] + b[f];
    out[m * 10 + f] = y * scale[0];
}

extern "C" void kernel_launch(void* const* d_in, const int* in_sizes, int n_in,
                              void* d_out, int out_size, void* d_ws,
                              size_t ws_size, hipStream_t stream) {
    const float* x   = (const float*)d_in[0];
    const float* cw0 = (const float*)d_in[1];
    const float* cp0 = (const float*)d_in[2];
    const float* cg0 = (const float*)d_in[3];
    const float* cb0 = (const float*)d_in[4];
    const float* cw1 = (const float*)d_in[5];
    const float* cp1 = (const float*)d_in[6];
    const float* cg1 = (const float*)d_in[7];
    const float* cb1 = (const float*)d_in[8];
    const float* fw0 = (const float*)d_in[9];
    const float* fp0 = (const float*)d_in[10];
    const float* fg0 = (const float*)d_in[11];
    const float* fb0 = (const float*)d_in[12];
    const float* fw1 = (const float*)d_in[13];
    const float* fp1 = (const float*)d_in[14];
    const float* fg1 = (const float*)d_in[15];
    const float* fb1 = (const float*)d_in[16];
    const float* scl = (const float*)d_in[17];
    float* out = (float*)d_out;

    char* ws = (char*)d_ws;
    int8_t*  t0   = (int8_t*)(ws + 0);
    int8_t*  a0p  = (int8_t*)(ws + 12845056);
    int8_t*  bw1  = (int8_t*)(ws + 27590656);
    int16_t* t1   = (int16_t*)(ws + 27721728);
    int8_t*  a1   = (int8_t*)(ws + 79101952);
    int*     tfc0 = (int*)(ws + 104792064);
    int8_t*  a2   = (int8_t*)(ws + 105840640);

    k_binw1<<<dim3((128 * 576 + 255) / 256), dim3(256), 0, stream>>>(cw1, bw1);
    k_conv0<<<dim3(784, 64), dim3(256), 0, stream>>>(x, cw0, t0);
    hipMemsetAsync(a0p, 0, (size_t)256 * 900 * 64, stream);
    k_bnbin0<<<dim3(64), dim3(256), 0, stream>>>(t0, cp0, cg0, cb0, a0p);
    k_conv1<<<dim3(12544), dim3(256), 0, stream>>>(a0p, bw1, t1);
    k_bnbin1<<<dim3(128), dim3(256), 0, stream>>>(t1, cp1, cg1, cb1, a1);
    hipMemsetAsync(tfc0, 0, 1024 * 256 * sizeof(int), stream);
    k_fc0<<<dim3(32, 8), dim3(256), 0, stream>>>(a1, fw0, tfc0);
    k_bnfc0<<<dim3(1024), dim3(256), 0, stream>>>(tfc0, fp0, fg0, fb0, a2);
    k_fc1<<<dim3(10), dim3(256), 0, stream>>>(a2, fw1, fp1, fg1, fb1, scl, out);
}
